// FaceEdgeVertexGCN_83932250898765
// MI455X (gfx1250) — compile-verified, matrix operand order checked by reading, not run on hardware
//
#include <hip/hip_runtime.h>
#include <hip/hip_bf16.h>
#include <stdint.h>

typedef __attribute__((ext_vector_type(16))) _Float16     v16h;
typedef __attribute__((ext_vector_type(8)))  float        v8f;
typedef __attribute__((ext_vector_type(4)))  unsigned int v4u;
typedef __attribute__((ext_vector_type(4)))  float        v4f;

#define LRELU(v) ((v) > 0.f ? (v) : 0.01f * (v))
#define ENC_NEG_INF ((int)0x80000000)

// Monotone float <-> ordered-int encoding (involution) so int atomicMax works.
__device__ __forceinline__ int f2ord(float f) {
    int i = __float_as_int(f);
    return (i >= 0) ? i : (i ^ 0x7fffffff);
}
__device__ __forceinline__ float ord2f(int i) {
    int j = (i >= 0) ? i : (i ^ 0x7fffffff);
    return __int_as_float(j);
}

__global__ void fill_i32(int* __restrict__ p, long long n, int v) {
    long long t = (long long)blockIdx.x * blockDim.x + threadIdx.x;
    if (t < n) p[t] = v;
}

// y = leaky_relu(x @ W + b); x:[rows,K], W:[K,32], y:[rows,32]
__global__ void embed_lrelu(const float* __restrict__ x, const float* __restrict__ W,
                            const float* __restrict__ b, float* __restrict__ out,
                            long long rows, int K) {
    long long t = (long long)blockIdx.x * blockDim.x + threadIdx.x;
    if (t >= rows * 32) return;
    long long r = t >> 5;
    int c = (int)(t & 31);
    float acc = b[c];
    for (int k = 0; k < K; ++k)
        acc = fmaf(x[r * K + k], W[(long long)k * 32 + c], acc);
    out[t] = LRELU(acc);
}

// Per (edge, batch): maxbuf[b,dst,c] = max(maxbuf, xdst[b,dst,c] - xsrc[b,src,c])
__global__ void scatter_max(const float* __restrict__ xsrc, const float* __restrict__ xdst,
                            const int* __restrict__ es, const int* __restrict__ ed,
                            int E, long long Ns, long long Nd, int* __restrict__ maxbuf) {
    int t = blockIdx.x * blockDim.x + threadIdx.x;
    if (t >= E * 4) return;
    int e = t >> 2, b = t & 3;
    long long s = es[e], d = ed[e];
    const v4f* ps = (const v4f*)(xsrc + ((long long)b * Ns + s) * 32);  // 128B aligned
    const v4f* pd = (const v4f*)(xdst + ((long long)b * Nd + d) * 32);
    int* pm = maxbuf + ((long long)b * Nd + d) * 32;
#pragma unroll
    for (int q = 0; q < 8; ++q) {
        v4f a = pd[q];
        v4f c = ps[q];
        atomicMax(&pm[4 * q + 0], f2ord(a.x - c.x));
        atomicMax(&pm[4 * q + 1], f2ord(a.y - c.y));
        atomicMax(&pm[4 * q + 2], f2ord(a.z - c.z));
        atomicMax(&pm[4 * q + 3], f2ord(a.w - c.w));
    }
}

// Load one 16-half WMMA operand fragment from LDS: two contiguous 16B chunks.
__device__ __forceinline__ v16h load_frag(const _Float16* base, int kb) {
    union { v16h h; v4u u[2]; } f;
    f.u[0] = *(const v4u*)(base + kb);        // K = kb .. kb+7
    f.u[1] = *(const v4u*)(base + 16 + kb);   // K = 16+kb .. 16+kb+7
    return f.h;
}

// out = xdst + leaky_relu([xdst | maxes] @ W + bias)
// rows = B*Nd, h:[rows,64], W:[64,32]. One wave per 16-row tile; WMMA f16->f32.
__global__ __launch_bounds__(256)
void mrconv_mlp_wmma(const float* __restrict__ xdst, const int* __restrict__ maxbuf,
                     const float* __restrict__ W, const float* __restrict__ bias,
                     float* __restrict__ out, int rows) {
    __shared__ _Float16 sWt[32 * 64];         // weights TRANSPOSED: [n][k], f16
    __shared__ float    sB[32];               // bias
    __shared__ _Float16 sH[8][16 * 64];       // per-wave staged h tiles [row][k], f16
    __shared__ float    sX[8][16 * 34];       // per-wave f32 residual tile, stride 34

    int tid = threadIdx.x;
    // Stage W transposed so B fragments are contiguous along K.
    for (int i = tid; i < 64 * 32; i += 256) {
        int k = i >> 5, n = i & 31;           // coalesced read of W[k][n]
        sWt[n * 64 + k] = (_Float16)W[i];
    }
    if (tid < 32) sB[tid] = bias[tid];

    int wave = tid >> 5;
    int lane = tid & 31;
    int m0 = (blockIdx.x * 8 + wave) * 16;
    int fullu = __builtin_amdgcn_readfirstlane((int)((m0 + 16) <= rows)); // wave-uniform

    // Stage h = [xdst(32) | decoded max(32)] as f16, plus f32 xdst residual copy.
    // Lanes 0-15: xdst column pairs; lanes 16-31: max column pairs. One b64 global
    // load + one b32 LDS store per row per lane, all immediate offsets.
    {
        unsigned int* sHu = (unsigned int*)&sH[wave][0];   // 16 rows x 32 uints
        bool loX = lane < 16;
        int cpair = 2 * (lane & 15);
        const int* gsrc = (loX ? (const int*)xdst : maxbuf) + (long long)m0 * 32 + cpair;
        if (fullu) {
#pragma unroll
            for (int row = 0; row < 16; ++row) {
                int2 raw = *(const int2*)(gsrc + row * 32);
                float v0, v1;
                if (loX) {
                    v0 = __int_as_float(raw.x);
                    v1 = __int_as_float(raw.y);
                } else {
                    v0 = (raw.x == ENC_NEG_INF) ? 0.f : ord2f(raw.x);
                    v1 = (raw.y == ENC_NEG_INF) ? 0.f : ord2f(raw.y);
                }
                if (loX) {
                    float2 tv; tv.x = v0; tv.y = v1;
                    *(float2*)&sX[wave][row * 34 + cpair] = tv;
                }
                union { _Float16 h[2]; unsigned int u; } pk;
                pk.h[0] = (_Float16)v0;
                pk.h[1] = (_Float16)v1;
                sHu[row * 32 + lane] = pk.u;
            }
        } else {
            for (int row = 0; row < 16; ++row) {
                float v0 = 0.f, v1 = 0.f;
                if (m0 + row < rows) {
                    int2 raw = *(const int2*)(gsrc + row * 32);
                    if (loX) {
                        v0 = __int_as_float(raw.x);
                        v1 = __int_as_float(raw.y);
                    } else {
                        v0 = (raw.x == ENC_NEG_INF) ? 0.f : ord2f(raw.x);
                        v1 = (raw.y == ENC_NEG_INF) ? 0.f : ord2f(raw.y);
                    }
                }
                if (loX) {
                    float2 tv; tv.x = v0; tv.y = v1;
                    *(float2*)&sX[wave][row * 34 + cpair] = tv;
                }
                union { _Float16 h[2]; unsigned int u; } pk;
                pk.h[0] = (_Float16)v0;
                pk.h[1] = (_Float16)v1;
                sHu[row * 32 + lane] = pk.u;
            }
        }
    }
    __syncthreads();

    // A fragments per ISA 16-bit 16x32 layout:
    // lanes 0-15: row M=lane, K = {0..7, 16..23}; lanes 16-31: same M, K = {8..15, 24..31}
    int mrow = lane & 15;
    int kb = (lane & 16) ? 8 : 0;
    const _Float16* hbase = &sH[wave][mrow * 64];
    v16h a0 = load_frag(hbase, kb);            // K-chunk 0 (xdst part)
    v16h a1 = load_frag(hbase + 32, kb);       // K-chunk 1 (maxes part)

    int hi8 = (lane & 16) ? 8 : 0;
    const float* resb = &sX[wave][hi8 * 34];   // residual rows for this half-wave
    int ncol0 = lane & 15;
#pragma unroll
    for (int nt = 0; nt < 2; ++nt) {
        int ncol = nt * 16 + ncol0;
        const _Float16* wbase = &sWt[ncol * 64];
        v16h b0 = load_frag(wbase, kb);        // K-chunk 0
        v16h b1 = load_frag(wbase + 32, kb);   // K-chunk 1

        v8f acc = {0.f, 0.f, 0.f, 0.f, 0.f, 0.f, 0.f, 0.f};
        acc = __builtin_amdgcn_wmma_f32_16x16x32_f16(false, a0, false, b0,
                                                     (short)0, acc, false, false);
        acc = __builtin_amdgcn_wmma_f32_16x16x32_f16(false, a1, false, b1,
                                                     (short)0, acc, false, false);

        // C/D layout: VGPR j -> M = j (lanes 0-15) or j+8 (lanes 16-31), N = lane&15
        float bb = sB[ncol];
        int rbase = m0 + hi8;
        float* po = out + (long long)rbase * 32 + ncol;  // one base, imm offsets below
        if (fullu) {
#pragma unroll
            for (int j = 0; j < 8; ++j) {
                float y = acc[j] + bb;
                po[j * 32] = resb[j * 34 + ncol] + LRELU(y);
            }
        } else {
#pragma unroll
            for (int j = 0; j < 8; ++j) {
                if (rbase + j < rows) {
                    float y = acc[j] + bb;
                    po[j * 32] = resb[j * 34 + ncol] + LRELU(y);
                }
            }
        }
    }
}

extern "C" void kernel_launch(void* const* d_in, const int* in_sizes, int n_in,
                              void* d_out, int out_size, void* d_ws, size_t ws_size,
                              hipStream_t stream) {
    (void)n_in; (void)out_size; (void)ws_size;
    const long long B = 4, C = 32;

    const float* x_f = (const float*)d_in[0];
    const float* x_e = (const float*)d_in[1];
    const float* x_v = (const float*)d_in[2];
    // d_in[3] = index_id (identity arange) -> not needed
    const int* fe = (const int*)d_in[4];
    const int* ev = (const int*)d_in[5];
    const int* ff = (const int*)d_in[6];
    const int* ef = (const int*)d_in[7];
    const int* ve = (const int*)d_in[8];
    const float* Wf = (const float*)d_in[9];   const float* bf = (const float*)d_in[10];
    const float* We = (const float*)d_in[11];  const float* be = (const float*)d_in[12];
    const float* Wv = (const float*)d_in[13];  const float* bv = (const float*)d_in[14];
    const float* W_f2e = (const float*)d_in[15]; const float* b_f2e = (const float*)d_in[16];
    const float* W_e2v = (const float*)d_in[17]; const float* b_e2v = (const float*)d_in[18];
    const float* W_ff  = (const float*)d_in[19]; const float* b_ff  = (const float*)d_in[20];
    const float* W_e2f = (const float*)d_in[21]; const float* b_e2f = (const float*)d_in[22];
    const float* W_v2e = (const float*)d_in[23]; const float* b_v2e = (const float*)d_in[24];

    long long Nf = in_sizes[0] / (B * 4);
    long long Ne = in_sizes[1] / (B * 6);
    long long Nv = in_sizes[2] / (B * 3);
    int E_fe = in_sizes[4] / 2;
    int E_ev = in_sizes[5] / 2;
    int E_ff = in_sizes[6] / 2;
    int E_ef = in_sizes[7] / 2;
    int E_ve = in_sizes[8] / 2;

    // Workspace layout (floats)
    float* xf0    = (float*)d_ws;
    float* xe0    = xf0 + B * Nf * C;
    float* xv0    = xe0 + B * Ne * C;
    float* xe_mid = xv0 + B * Nv * C;
    float* xf_mid = xe_mid + B * Ne * C;
    int*   maxbuf = (int*)(xf_mid + B * Nf * C);   // reused, sized B*max(Nd)*C

    float* out_f = (float*)d_out;
    float* out_e = out_f + B * Nf * C;
    float* out_v = out_e + B * Ne * C;

    auto cdiv = [](long long a, long long b) { return (unsigned)((a + b - 1) / b); };

    // Initial embeddings
    embed_lrelu<<<cdiv(B * Nf * C, 256), 256, 0, stream>>>(x_f, Wf, bf, xf0, B * Nf, 4);
    embed_lrelu<<<cdiv(B * Ne * C, 256), 256, 0, stream>>>(x_e, We, be, xe0, B * Ne, 6);
    embed_lrelu<<<cdiv(B * Nv * C, 256), 256, 0, stream>>>(x_v, Wv, bv, xv0, B * Nv, 3);

    auto mrconv = [&](const float* xsrc, const float* xd, const int* edges, int E,
                      long long Ns, long long Nd, const float* Wm, const float* bm,
                      float* outp) {
        long long cnt = B * Nd * C;
        fill_i32<<<cdiv(cnt, 256), 256, 0, stream>>>(maxbuf, cnt, ENC_NEG_INF);
        scatter_max<<<cdiv((long long)E * B, 256), 256, 0, stream>>>(
            xsrc, xd, edges, edges + E, E, Ns, Nd, maxbuf);
        int rows = (int)(B * Nd);
        mrconv_mlp_wmma<<<cdiv(rows, 128), 256, 0, stream>>>(xd, maxbuf, Wm, bm, outp, rows);
    };

    mrconv(xf0,    xe0,    fe, E_fe, Nf, Ne, W_f2e, b_f2e, xe_mid); // F2E
    mrconv(xe0,    xv0,    ev, E_ev, Ne, Nv, W_e2v, b_e2v, out_v);  // E2V -> final xv
    mrconv(xf0,    xf0,    ff, E_ff, Nf, Nf, W_ff,  b_ff,  xf_mid); // F2F
    mrconv(xe_mid, xf_mid, ef, E_ef, Ne, Nf, W_e2f, b_e2f, out_f);  // E2F -> final xf
    mrconv(out_v,  xe_mid, ve, E_ve, Nv, Ne, W_v2e, b_v2e, out_e);  // V2E -> final xe
}
